// OnDeviceBeamSearch_5952824672597
// MI455X (gfx1250) — matrix operand order repaired and checked by reading, not compiled, not run on hardware
//
#include <hip/hip_runtime.h>
#include <hip/hip_bf16.h>
#include <math.h>

// Problem constants (from the reference)
#define Bq 64
#define Kq 8
#define Lq 256
#define Vq 50257
#define Dq 1024
#define Mq (Bq*Kq)          // 512 rows in the GEMM
#define EOSq 50256
#define LNCq (-1000000000.0f)
#define NEG_INF (-3.0e38f)

// Output layout (flat f32): [B*L gen_tokens][1 done][B*K next_log_probs][B*K next_running_log_probs]
#define OUT_DONE (Bq*Lq)
#define OUT_NLP  (Bq*Lq + 1)
#define OUT_NRLP (Bq*Lq + 1 + Bq*Kq)

typedef __attribute__((ext_vector_type(16))) __bf16 v16bf;
typedef __attribute__((ext_vector_type(8)))  float  v8f;

static __device__ __forceinline__ unsigned short f2bf(float f) {
  // round-to-nearest-even f32 -> bf16
  unsigned int u = __float_as_uint(f);
  u += 0x7fffu + ((u >> 16) & 1u);
  return (unsigned short)(u >> 16);
}

union AFrag { v16bf v; unsigned short u[16]; uint4 q[2]; };

// ---------------------------------------------------------------------------
// K1: gather token ids and embed rows, convert activations to bf16 (512x1024)
// ---------------------------------------------------------------------------
__global__ void k_gather_embed(const int* __restrict__ input_ids,
                               const int* __restrict__ run_seq,
                               const float* __restrict__ embed,
                               const int* __restrict__ step_p,
                               unsigned short* __restrict__ Xb) {
  int bk = blockIdx.x;
  int step = *step_p;
  // running_sequences has position 0 replaced by input_ids
  int tok = (step - 1 == 0) ? input_ids[bk] : run_seq[bk * Lq + (step - 1)];
  const float* src = embed + (size_t)tok * Dq;
  for (int d = threadIdx.x; d < Dq; d += blockDim.x)
    Xb[bk * Dq + d] = f2bf(src[d]);
}

// ---------------------------------------------------------------------------
// K2: logits = Xb (512x1024 bf16) @ W (1024x50257 f32 -> bf16 in LDS).
// One block per 16-column N tile; 8 waves x 4 M-tiles cover all 512 rows, so
// each W tile is read from HBM exactly once (roofline-optimal: ~206 MB once).
// LDS tile is stored transposed [n][k] so each lane's B fragment (column
// N = lane&15, K = kbA..kbA+7 and kbB..kbB+7) is two aligned ds_load_b128.
// ---------------------------------------------------------------------------
__global__ __launch_bounds__(256) void k_gemm(const unsigned short* __restrict__ Xb,
                                              const float* __restrict__ W,
                                              float* __restrict__ logits) {
  __shared__ unsigned short ldsB[16][40];   // [n][k], row padded to 80B
  const int n0   = blockIdx.x * 16;
  const int tid  = threadIdx.x;
  const int lane = tid & 31;
  const int wave = tid >> 5;
  const int nl   = lane & 15;
  const int hg   = lane >> 4;               // lane half-group
  const int kbA  = hg * 8;                  // A fragment K base (ISA 7.12.2)
  const int kbB  = kbA + 16;                // B fragment K base

  v8f acc[4];
  v8f zero = {0.f, 0.f, 0.f, 0.f, 0.f, 0.f, 0.f, 0.f};
#pragma unroll
  for (int i = 0; i < 4; ++i) acc[i] = zero;

  for (int k0 = 0; k0 < Dq; k0 += 32) {
    __syncthreads();
    // block-cooperative load of 32x16 W tile -> transposed bf16 LDS
#pragma unroll
    for (int e = tid; e < 512; e += 256) {
      int kk = e >> 4, nn = e & 15;
      int n = n0 + nn;
      float w = (n < Vq) ? W[(size_t)(k0 + kk) * Vq + n] : 0.f;
      ldsB[nn][kk] = f2bf(w);
      if (k0 + 32 < Dq)   // gfx1250 global_prefetch_b8 for next K tile
        __builtin_prefetch(&W[(size_t)(k0 + 32 + kk) * Vq + n0 + nn], 0, 3);
    }
    __syncthreads();

    // B fragment: lane holds column N = nl, K = kbA+0..7 and kbA+16..23 (matches A)
    AFrag bfr;
    bfr.q[0] = *(const uint4*)&ldsB[nl][kbA];
    bfr.q[1] = *(const uint4*)&ldsB[nl][kbB];

    // Preload all 4 A fragments so the 8 global_load_b128 overlap the WMMAs
    AFrag af[4];
#pragma unroll
    for (int i = 0; i < 4; ++i) {
      int row = (wave * 4 + i) * 16 + nl;   // A: M = lane&15 within tile
      const unsigned short* ap = Xb + (size_t)row * Dq + k0 + kbA;
      af[i].q[0] = *(const uint4*)(ap);       // K = kbA+0..7
      af[i].q[1] = *(const uint4*)(ap + 16);  // K = kbA+16..23
    }
#pragma unroll
    for (int i = 0; i < 4; ++i)
      acc[i] = __builtin_amdgcn_wmma_f32_16x16x32_bf16(
          false, af[i].v, false, bfr.v, (short)0, acc[i], false, false);
  }

  // C/D layout: VGPR r -> M = r + 8*hg, N = lane&15
  int col = n0 + nl;
  if (col < Vq) {
#pragma unroll
    for (int i = 0; i < 4; ++i) {
      int mbase = (wave * 4 + i) * 16 + hg * 8;
#pragma unroll
      for (int r = 0; r < 8; ++r)
        logits[(size_t)(mbase + r) * Vq + col] = acc[i][r];
    }
  }
}

// ---------------------------------------------------------------------------
// K3: per-row logsumexp (for log_softmax). One block per row.
// ---------------------------------------------------------------------------
__global__ __launch_bounds__(256) void k_lse(const float* __restrict__ logits,
                                             float* __restrict__ lse) {
  __shared__ float red[256];
  int bk = blockIdx.x, tid = threadIdx.x;
  const float* row = logits + (size_t)bk * Vq;
  float m = NEG_INF;
  for (int v = tid; v < Vq; v += 256) m = fmaxf(m, row[v]);
  red[tid] = m; __syncthreads();
  for (int off = 128; off > 0; off >>= 1) {
    if (tid < off) red[tid] = fmaxf(red[tid], red[tid + off]);
    __syncthreads();
  }
  float gm = red[0]; __syncthreads();
  float s = 0.f;
  for (int v = tid; v < Vq; v += 256) s += expf(row[v] - gm);
  red[tid] = s; __syncthreads();
  for (int off = 128; off > 0; off >>= 1) {
    if (tid < off) red[tid] += red[tid + off];
    __syncthreads();
  }
  if (tid == 0) lse[bk] = gm + logf(red[0]);
}

// ---------------------------------------------------------------------------
// K4: stable top-16 of (logits - lse + running_log_prob) over K*V per batch.
// Per-thread sorted-register top-16 insertion, then 16-round LDS tournament.
// Ties resolved toward the lower flat index (matches jax.lax.top_k).
// ---------------------------------------------------------------------------
__global__ __launch_bounds__(256) void k_top16(const float* __restrict__ logits,
                                               const float* __restrict__ lse,
                                               const float* __restrict__ rlp,
                                               float* __restrict__ t16s,
                                               int* __restrict__ t16i) {
  __shared__ float shS[256 * 16];
  __shared__ int   shI[256 * 16];
  __shared__ float rs[256];
  __shared__ int   ri[256];
  __shared__ int   rslot[256];
  int b = blockIdx.x, tid = threadIdx.x;

  float ls[16]; int li[16];
#pragma unroll
  for (int j = 0; j < 16; ++j) { ls[j] = NEG_INF; li[j] = 0x7fffffff; }

  for (int k = 0; k < Kq; ++k) {
    int bk = b * Kq + k;
    const float* row = logits + (size_t)bk * Vq;
    float bias = rlp[bk] - lse[bk];
    int base = k * Vq;
    for (int v = tid; v < Vq; v += 256) {
      float sc = row[v] + bias;
      int ix = base + v;
      if (sc > ls[15] || (sc == ls[15] && ix < li[15])) {
        ls[15] = sc; li[15] = ix;
#pragma unroll
        for (int p = 15; p > 0; --p) {
          bool mv = (ls[p] > ls[p-1]) || (ls[p] == ls[p-1] && li[p] < li[p-1]);
          if (mv) {
            float tf = ls[p]; ls[p] = ls[p-1]; ls[p-1] = tf;
            int   ti = li[p]; li[p] = li[p-1]; li[p-1] = ti;
          }
        }
      }
    }
  }
#pragma unroll
  for (int j = 0; j < 16; ++j) { shS[tid*16+j] = ls[j]; shI[tid*16+j] = li[j]; }
  __syncthreads();

  for (int r = 0; r < 16; ++r) {
    float bs = NEG_INF; int bi = 0x7fffffff, bsl = 0;
#pragma unroll
    for (int j = 0; j < 16; ++j) {
      float s2 = shS[tid*16+j]; int i2 = shI[tid*16+j];
      if (s2 > bs || (s2 == bs && i2 < bi)) { bs = s2; bi = i2; bsl = tid*16+j; }
    }
    rs[tid] = bs; ri[tid] = bi; rslot[tid] = bsl;
    __syncthreads();
    for (int off = 128; off > 0; off >>= 1) {
      if (tid < off) {
        if (rs[tid+off] > rs[tid] ||
            (rs[tid+off] == rs[tid] && ri[tid+off] < ri[tid])) {
          rs[tid] = rs[tid+off]; ri[tid] = ri[tid+off]; rslot[tid] = rslot[tid+off];
        }
      }
      __syncthreads();
    }
    if (tid == 0) {
      t16s[b*16 + r] = rs[0];
      t16i[b*16 + r] = ri[0];
      shS[rslot[0]] = NEG_INF;
      shI[rslot[0]] = 0x7fffffff;
    }
    __syncthreads();
  }
}

// ---------------------------------------------------------------------------
// K5: per-batch beam bookkeeping + winning sequence copy-out.
// ---------------------------------------------------------------------------
__global__ __launch_bounds__(256) void k_finalize(
    const int* __restrict__ input_ids, const int* __restrict__ seq_state,
    const int* __restrict__ run_seq, const float* __restrict__ lp_state,
    const float* __restrict__ rlp_state, const int* __restrict__ fin_state,
    const float* __restrict__ t16s, const int* __restrict__ t16i,
    const int* __restrict__ step_p, float* __restrict__ out,
    int* __restrict__ impr) {
  __shared__ int sh_mode, sh_beam, sh_tok;
  int b = blockIdx.x, tid = threadIdx.x;
  int step = *step_p;

  if (tid == 0) {
    float st = (float)step;
    float s[16]; int beam[16], tok[16]; bool fin[16];
    for (int i = 0; i < 16; ++i) {
      s[i] = t16s[b*16 + i];
      int ix = t16i[b*16 + i];
      beam[i] = ix / Vq; tok[i] = ix % Vq; fin[i] = (tok[i] == EOSq);
    }
    // running scores with EOS knocked out; stable top-8 -> next running beams
    float run[16];
    for (int i = 0; i < 16; ++i) run[i] = s[i] + (fin[i] ? LNCq : 0.f);
    bool used[16]; for (int i = 0; i < 16; ++i) used[i] = false;
    int sel0 = 0;
    for (int j = 0; j < Kq; ++j) {
      int best = -1;
      for (int i = 0; i < 16; ++i) {
        if (used[i]) continue;
        if (best < 0 || run[i] > run[best]) best = i;  // strict > keeps low idx on ties
      }
      used[best] = true;
      if (j == 0) sel0 = best;
      out[OUT_NRLP + b*Kq + j] = run[best];
    }
    // length-penalized scores; unfinished candidates knocked out (full=false)
    float mg[24]; bool mf[24];
    for (int j = 0; j < Kq; ++j) { mg[j] = lp_state[b*Kq + j]; mf[j] = (fin_state[b*Kq + j] != 0); }
    for (int i = 0; i < 16; ++i) { mg[Kq+i] = s[i]/st + (fin[i] ? 0.f : LNCq); mf[Kq+i] = fin[i]; }
    bool used2[24]; for (int i = 0; i < 24; ++i) used2[i] = false;
    int msel0 = 0; bool anyfin = false;
    for (int j = 0; j < Kq; ++j) {
      int best = -1;
      for (int i = 0; i < 24; ++i) {
        if (used2[i]) continue;
        if (best < 0 || mg[i] > mg[best]) best = i;
      }
      used2[best] = true;
      if (j == 0) msel0 = best;
      out[OUT_NLP + b*Kq + j] = mg[best];
      anyfin = anyfin || mf[best];
    }
    // improvement flag for the global done reduction
    float wmin = lp_state[b*Kq];
    for (int j = 1; j < Kq; ++j) wmin = fminf(wmin, lp_state[b*Kq + j]);
    float worst_fin = anyfin ? wmin : LNCq;
    float best_running = rlp_state[b*Kq] / st;
    impr[b] = (best_running > worst_fin) ? 1 : 0;
    // which sequence row 0 to emit
    if (anyfin) {
      if (msel0 < Kq) { sh_mode = 0; sh_beam = msel0; sh_tok = 0; }
      else { int i = msel0 - Kq; sh_mode = 1; sh_beam = beam[i]; sh_tok = tok[i]; }
    } else {
      sh_mode = 1; sh_beam = beam[sel0]; sh_tok = tok[sel0];
    }
  }
  __syncthreads();
  int mode = sh_mode, bm = sh_beam, tk = sh_tok;
  for (int t = tid; t < Lq; t += 256) {
    int val;
    if (mode == 0) {
      val = seq_state[(b*Kq + bm) * Lq + t];
    } else {
      val = run_seq[(b*Kq + bm) * Lq + t];
      if (t == 0)    val = input_ids[b*Kq + bm];
      if (t == step) val = tk;
    }
    out[b*Lq + t] = (float)val;
  }
}

// ---------------------------------------------------------------------------
// K6: global done flag.
// ---------------------------------------------------------------------------
__global__ void k_done(const int* __restrict__ impr,
                       const int* __restrict__ step_p,
                       float* __restrict__ out) {
  __shared__ int any;
  if (threadIdx.x == 0) any = 0;
  __syncthreads();
  if (threadIdx.x < Bq && impr[threadIdx.x]) atomicOr(&any, 1);
  __syncthreads();
  if (threadIdx.x == 0) {
    int step = *step_p;
    bool cont = (step < Lq) && (any != 0);   // still_open is always true here
    out[OUT_DONE] = cont ? 0.f : 1.f;
  }
}

extern "C" void kernel_launch(void* const* d_in, const int* in_sizes, int n_in,
                              void* d_out, int out_size, void* d_ws, size_t ws_size,
                              hipStream_t stream) {
  const int*   input_ids = (const int*)  d_in[0];
  const int*   seq_state = (const int*)  d_in[1];
  const int*   run_seq   = (const int*)  d_in[2];
  const float* lp        = (const float*)d_in[3];
  const float* rlp       = (const float*)d_in[4];
  const int*   finst     = (const int*)  d_in[5];
  const float* embed     = (const float*)d_in[6];
  const float* W         = (const float*)d_in[7];
  const int*   step_p    = (const int*)  d_in[8];
  float* out = (float*)d_out;

  // Workspace layout (~104 MB): Xbf16 | logits f32 | lse | top16 | flags
  char* ws = (char*)d_ws;
  size_t off = 0;
  unsigned short* Xb = (unsigned short*)(ws + off); off += (size_t)Mq * Dq * 2;
  off = (off + 255) & ~(size_t)255;
  float* logits = (float*)(ws + off); off += (size_t)Mq * Vq * 4;
  off = (off + 255) & ~(size_t)255;
  float* lse  = (float*)(ws + off); off += (size_t)Mq * 4;
  float* t16s = (float*)(ws + off); off += (size_t)Bq * 16 * 4;
  int*   t16i = (int*)  (ws + off); off += (size_t)Bq * 16 * 4;
  int*   impr = (int*)  (ws + off); off += (size_t)Bq * 4;

  k_gather_embed<<<Mq, 256, 0, stream>>>(input_ids, run_seq, embed, step_p, Xb);
  int ntiles = (Vq + 15) / 16;                 // 3142 column tiles
  k_gemm<<<ntiles, 256, 0, stream>>>(Xb, W, logits);
  k_lse<<<Mq, 256, 0, stream>>>(logits, lse);
  k_top16<<<Bq, 256, 0, stream>>>(logits, lse, rlp, t16s, t16i);
  k_finalize<<<Bq, 256, 0, stream>>>(input_ids, seq_state, run_seq, lp, rlp,
                                     finst, t16s, t16i, step_p, out, impr);
  k_done<<<1, 64, 0, stream>>>(impr, step_p, out);
}
